// Net_90537910600276
// MI455X (gfx1250) — matrix operand order repaired and checked by reading, not compiled, not run on hardware
//
#include <hip/hip_runtime.h>
#include <hip/hip_bf16.h>

typedef __attribute__((ext_vector_type(16))) _Float16 v16h;
typedef __attribute__((ext_vector_type(8)))  _Float16 v8h;
typedef __attribute__((ext_vector_type(8)))  float    v8f;

// ---------------- problem constants ----------------
#define N_IN   784
#define N_INP  800      // padded K for layer0 GEMM (multiple of 32)
#define N_HID  2048
#define N_OUT  10
#define N_OUTP 16       // padded N for last GEMM
#define BATCH  128
#define TSTEPS 32
#define BETA   0.9f

// ---------------- workspace layout (bytes, 256-aligned) ----------------
#define OFF_W0H   ((size_t)0)                          // 2048*800*2   = 3276800
#define OFF_W1H   (OFF_W0H + 3276800)                  // 2048*2048*2  = 8388608
#define OFF_W2H   (OFF_W1H + 8388608)
#define OFF_W3H   (OFF_W2H + 8388608)
#define OFF_W4H   (OFF_W3H + 8388608)                  // 16*2048*2    = 65536
#define OFF_MEM0  (OFF_W4H + 65536)                    // 128*784*4    = 401408
#define OFF_MEM1  (OFF_MEM0 + 401408)                  // 128*2048*4   = 1048576
#define OFF_MEM2  (OFF_MEM1 + 1048576)
#define OFF_MEM3  (OFF_MEM2 + 1048576)
#define OFF_MEM4  (OFF_MEM3 + 1048576)
#define OFF_MEM5  (OFF_MEM4 + 1048576)                 // 128*16*4     = 8192
#define OFF_SPK0  (OFF_MEM5 + 8192)                    // 128*800*2    = 204800
#define OFF_SPK1  (OFF_SPK0 + 204800)                  // 128*2048*2   = 524288
#define OFF_SPK2  (OFF_SPK1 + 524288)
#define OFF_SPK3  (OFF_SPK2 + 524288)
#define OFF_SPK4  (OFF_SPK3 + 524288)
#define OFF_SUMP  (OFF_SPK4 + 524288)                  // 32*5*1024*4  = 655360
#define OFF_SQP   (OFF_SUMP + 655360)
#define OFF_CNTP  (OFF_SQP  + 655360)                  // 32*6*1024*4  = 786432
#define OFF_VSTEP (OFF_CNTP + 786432)                  // 160*4
#define ZERO1_OFF  OFF_MEM0                            // mems + spk0 (pad tail)
#define ZERO1_SIZE (OFF_SPK1 - OFF_MEM0)
#define ZERO2_OFF  OFF_SUMP                            // partial arrays
#define ZERO2_SIZE (OFF_VSTEP + 1024 - OFF_SUMP)

// ---------------- fp32 -> fp16 weight convert with zero padding ----------------
__global__ __launch_bounds__(256) void cvt_w(const float* __restrict__ W,
                                             _Float16* __restrict__ Wh,
                                             int N, int K, int Npad, int Kpad) {
    int total = Npad * Kpad;
    for (int idx = blockIdx.x * 256 + threadIdx.x; idx < total; idx += gridDim.x * 256) {
        int n = idx / Kpad;
        int k = idx - n * Kpad;
        Wh[idx] = (n < N && k < K) ? (_Float16)W[n * K + k] : (_Float16)0.0f;
    }
}

// ---------------- LIF layer 0 (no GEMM; input injected every step) ----------------
__global__ __launch_bounds__(128) void lif0_kernel(const float* __restrict__ inp,
                                                   float* __restrict__ mem,
                                                   _Float16* __restrict__ spk,
                                                   float* __restrict__ sumP,
                                                   float* __restrict__ sqP,
                                                   float* __restrict__ cntP) {
    const int tid = blockIdx.x * 128 + threadIdx.x;
    const int stride = gridDim.x * 128;
    float sSum = 0.f, sSq = 0.f, sCnt = 0.f;
    for (int idx = tid; idx < BATCH * N_IN; idx += stride) {
        int m = idx / N_IN;
        int k = idx - m * N_IN;
        float mprev = mem[idx];
        float reset = (mprev > 1.0f) ? 1.0f : 0.0f;
        float mnew  = BETA * mprev + inp[idx] - reset;
        float spike = (mnew > 1.0f) ? 1.0f : 0.0f;
        mem[idx] = mnew;
        spk[m * N_INP + k] = (_Float16)spike;
        sSum += mnew; sSq += mnew * mnew; sCnt += spike;
    }
    #pragma unroll
    for (int off = 16; off > 0; off >>= 1) {
        sSum += __shfl_xor(sSum, off, 32);
        sSq  += __shfl_xor(sSq,  off, 32);
        sCnt += __shfl_xor(sCnt, off, 32);
    }
    if ((threadIdx.x & 31) == 0) {
        int slot = blockIdx.x * 4 + (threadIdx.x >> 5);   // 0..1023
        sumP[slot] = sSum; sqP[slot] = sSq; cntP[slot] = sCnt;
    }
}

// ---------------- fused GEMM (spk @ W^T + b) + LIF epilogue ----------------
// A: [BATCH, K] f16 spikes (K padded, zero tail). Wh: [Npad, K] f16 row-major.
// One 16x16 output tile per wave; compile-time K so the v_wmma_f32_16x16x32_f16
// K-loop has a constant trip count (800/32=25, 2048/32=64) and unrolls deeply.
template <int K, bool LAST>
__global__ __launch_bounds__(128) void gemm_lif_kernel(
    const _Float16* __restrict__ A, const _Float16* __restrict__ Wh,
    const float* __restrict__ bias, float* __restrict__ mem,
    _Float16* __restrict__ spk_out,          // hidden layers: next-layer spikes
    float* __restrict__ spk_final,           // last layer: d_out slice
    int Nreal, int Npad,
    float* __restrict__ sumP, float* __restrict__ sqP, float* __restrict__ cntP) {

    const int lane = threadIdx.x & 31;
    const int wave = threadIdx.x >> 5;
    const int tile = blockIdx.x * 4 + wave;       // grid sized so every tile valid
    const int m0 = (tile & 7) * 16;               // 8 M-tiles (BATCH=128)
    const int n0 = (tile >> 3) * 16;

    // A operand, contiguous K order used here (same as B): lanes 0-15 -> K 0..15 ; lanes 16-31 -> K 16..31
    const _Float16* aPtr = A + (m0 + (lane & 15)) * K + ((lane >> 4) * 16);
    // B operand, same contiguous K order as A: lanes 0-15 -> K 0..15 ; lanes 16-31 -> K 16..31 (N = lane&15)
    const _Float16* bPtr = Wh + (n0 + (lane & 15)) * K + ((lane >> 4) * 16);

    v8f c = {};
    #pragma unroll 4
    for (int k0 = 0; k0 < K; k0 += 32) {
        v8h alo = *(const v8h*)(aPtr + k0);
        v8h ahi = *(const v8h*)(aPtr + k0 + 8);
        v16h a;
        #pragma unroll
        for (int i = 0; i < 8; ++i) { a[i] = alo[i]; a[i + 8] = ahi[i]; }
        v16h b = *(const v16h*)(bPtr + k0);
        c = __builtin_amdgcn_wmma_f32_16x16x32_f16(false, a, false, b,
                                                   (short)0, c, false, false);
    }

    // C/D layout: VGPR j, lane l -> M = m0 + j + 8*(l>=16), N = n0 + (l&15)
    const int nn    = n0 + (lane & 15);
    const int mBase = m0 + ((lane >> 4) << 3);
    const float bv  = (nn < Nreal) ? bias[nn] : 0.0f;

    float sSum = 0.f, sSq = 0.f, sCnt = 0.f;
    #pragma unroll
    for (int j = 0; j < 8; ++j) {
        const int mm = mBase + j;
        float* mp = mem + mm * Npad + nn;
        float mprev = *mp;
        float reset = (mprev > 1.0f) ? 1.0f : 0.0f;
        float mnew  = BETA * mprev + (c[j] + bv) - reset;
        float spike = (mnew > 1.0f) ? 1.0f : 0.0f;
        *mp = mnew;
        if (LAST) {
            if (nn < Nreal) spk_final[mm * Nreal + nn] = spike;
        } else {
            spk_out[mm * Npad + nn] = (_Float16)spike;
        }
        sSum += mnew; sSq += mnew * mnew; sCnt += spike;
    }
    #pragma unroll
    for (int off = 16; off > 0; off >>= 1) {
        sSum += __shfl_xor(sSum, off, 32);
        sSq  += __shfl_xor(sSq,  off, 32);
        sCnt += __shfl_xor(sCnt, off, 32);
    }
    if (lane == 0) {
        if (!LAST) { sumP[tile] = sSum; sqP[tile] = sSq; }
        cntP[tile] = sCnt;
    }
}

// ---------------- per-(step,layer) unbiased variance from partials ----------------
__global__ __launch_bounds__(256) void var_step_kernel(const float* __restrict__ sumP,
                                                       const float* __restrict__ sqP,
                                                       float* __restrict__ varStep) {
    __shared__ float sS[256], sQ[256];
    const int b = blockIdx.x;            // 0..159  (t*5 + l)
    const int l = b % 5;
    const float* sp = sumP + b * 1024;
    const float* qp = sqP  + b * 1024;
    const int tid = threadIdx.x;
    float s = 0.f, q = 0.f;
    for (int i = tid; i < 1024; i += 256) { s += sp[i]; q += qp[i]; }
    sS[tid] = s; sQ[tid] = q; __syncthreads();
    for (int off = 128; off > 0; off >>= 1) {
        if (tid < off) { sS[tid] += sS[tid + off]; sQ[tid] += sQ[tid + off]; }
        __syncthreads();
    }
    if (tid == 0) {
        const float n = 128.0f * ((l == 0) ? 784.0f : 2048.0f);
        varStep[b] = (sQ[0] - sS[0] * sS[0] / n) / (n - 1.0f);
    }
}

// ---------------- spike-count totals (exact: integer-valued floats) ----------------
__global__ __launch_bounds__(256) void count_kernel(const float* __restrict__ cntP,
                                                    float* __restrict__ out) {
    __shared__ float sC[256];
    const int l = blockIdx.x;            // 0..5
    const int tid = threadIdx.x;
    float s = 0.f;
    for (int t = 0; t < TSTEPS; ++t)
        for (int i = tid; i < 1024; i += 256)
            s += cntP[(t * 6 + l) * 1024 + i];
    sC[tid] = s; __syncthreads();
    for (int off = 128; off > 0; off >>= 1) {
        if (tid < off) sC[tid] += sC[tid + off];
        __syncthreads();
    }
    if (tid == 0) out[l] = sC[0] / ((float)TSTEPS * (float)BATCH * (float)BATCH);
}

// ---------------- mean over time of variances ----------------
__global__ void var_mean_kernel(const float* __restrict__ varStep, float* __restrict__ out) {
    const int l = threadIdx.x;
    if (l < 5) {
        float s = 0.f;
        for (int t = 0; t < TSTEPS; ++t) s += varStep[t * 5 + l];
        out[l] = s / (float)TSTEPS;
    }
}

extern "C" void kernel_launch(void* const* d_in, const int* in_sizes, int n_in,
                              void* d_out, int out_size, void* d_ws, size_t ws_size,
                              hipStream_t stream) {
    const float* inp = (const float*)d_in[0];
    const float* W[5] = {(const float*)d_in[1], (const float*)d_in[2], (const float*)d_in[3],
                         (const float*)d_in[4], (const float*)d_in[5]};
    const float* B[5] = {(const float*)d_in[6], (const float*)d_in[7], (const float*)d_in[8],
                         (const float*)d_in[9], (const float*)d_in[10]};
    float* out = (float*)d_out;
    char* ws = (char*)d_ws;

    _Float16* W0h = (_Float16*)(ws + OFF_W0H);
    _Float16* W1h = (_Float16*)(ws + OFF_W1H);
    _Float16* W2h = (_Float16*)(ws + OFF_W2H);
    _Float16* W3h = (_Float16*)(ws + OFF_W3H);
    _Float16* W4h = (_Float16*)(ws + OFF_W4H);
    float* mem0 = (float*)(ws + OFF_MEM0);
    float* mem1 = (float*)(ws + OFF_MEM1);
    float* mem2 = (float*)(ws + OFF_MEM2);
    float* mem3 = (float*)(ws + OFF_MEM3);
    float* mem4 = (float*)(ws + OFF_MEM4);
    float* mem5 = (float*)(ws + OFF_MEM5);
    _Float16* spk0 = (_Float16*)(ws + OFF_SPK0);
    _Float16* spk1 = (_Float16*)(ws + OFF_SPK1);
    _Float16* spk2 = (_Float16*)(ws + OFF_SPK2);
    _Float16* spk3 = (_Float16*)(ws + OFF_SPK3);
    _Float16* spk4 = (_Float16*)(ws + OFF_SPK4);
    float* sumP    = (float*)(ws + OFF_SUMP);
    float* sqP     = (float*)(ws + OFF_SQP);
    float* cntP    = (float*)(ws + OFF_CNTP);
    float* varStep = (float*)(ws + OFF_VSTEP);

    // zero membranes + spk0 pad tail, and all partial-sum slots (capture-safe async memset)
    hipMemsetAsync(ws + ZERO1_OFF, 0, ZERO1_SIZE, stream);
    hipMemsetAsync(ws + ZERO2_OFF, 0, ZERO2_SIZE, stream);

    // one-time fp32 -> fp16 weight conversion (zero-padded)
    cvt_w<<<512, 256, 0, stream>>>(W[0], W0h, N_HID, N_IN,  N_HID,  N_INP);
    cvt_w<<<512, 256, 0, stream>>>(W[1], W1h, N_HID, N_HID, N_HID,  N_HID);
    cvt_w<<<512, 256, 0, stream>>>(W[2], W2h, N_HID, N_HID, N_HID,  N_HID);
    cvt_w<<<512, 256, 0, stream>>>(W[3], W3h, N_HID, N_HID, N_HID,  N_HID);
    cvt_w<<<64,  256, 0, stream>>>(W[4], W4h, N_OUT, N_HID, N_OUTP, N_HID);

    for (int t = 0; t < TSTEPS; ++t) {
        float* su = sumP + (size_t)(t * 5) * 1024;
        float* sq = sqP  + (size_t)(t * 5) * 1024;
        float* cn = cntP + (size_t)(t * 6) * 1024;

        lif0_kernel<<<256, 128, 0, stream>>>(inp, mem0, spk0, su, sq, cn);

        gemm_lif_kernel<N_INP, false><<<256, 128, 0, stream>>>(
            spk0, W0h, B[0], mem1, spk1, nullptr, N_HID, N_HID,
            su + 1024, sq + 1024, cn + 1024);
        gemm_lif_kernel<N_HID, false><<<256, 128, 0, stream>>>(
            spk1, W1h, B[1], mem2, spk2, nullptr, N_HID, N_HID,
            su + 2 * 1024, sq + 2 * 1024, cn + 2 * 1024);
        gemm_lif_kernel<N_HID, false><<<256, 128, 0, stream>>>(
            spk2, W2h, B[2], mem3, spk3, nullptr, N_HID, N_HID,
            su + 3 * 1024, sq + 3 * 1024, cn + 3 * 1024);
        gemm_lif_kernel<N_HID, false><<<256, 128, 0, stream>>>(
            spk3, W3h, B[3], mem4, spk4, nullptr, N_HID, N_HID,
            su + 4 * 1024, sq + 4 * 1024, cn + 4 * 1024);
        // last layer: 8 tiles (8 M x 1 N), spikes straight into d_out[5 + t*1280 ...]
        gemm_lif_kernel<N_HID, true><<<2, 128, 0, stream>>>(
            spk4, W4h, B[4], mem5, nullptr, out + 5 + t * (BATCH * N_OUT),
            N_OUT, N_OUTP, nullptr, nullptr, cn + 5 * 1024);
    }

    var_step_kernel<<<160, 256, 0, stream>>>(sumP, sqP, varStep);
    count_kernel<<<6, 256, 0, stream>>>(cntP, out + 5 + TSTEPS * BATCH * N_OUT);
    var_mean_kernel<<<1, 32, 0, stream>>>(varStep, out);
}
